// PhaseAssociativeMemory_35278861369599
// MI455X (gfx1250) — matrix operand order repaired and checked by reading, not compiled, not run on hardware
//
#include <hip/hip_runtime.h>

// ---------------------------------------------------------------- types
typedef __bf16 bf16_t;
typedef __bf16 v16bf __attribute__((ext_vector_type(16)));
typedef __bf16 v8bf  __attribute__((ext_vector_type(8)));
typedef __bf16 v2bf  __attribute__((ext_vector_type(2)));
typedef float  v8f   __attribute__((ext_vector_type(8)));

// ---------------------------------------------------------------- consts
constexpr int BSROWS = 4096;   // B * S
constexpr int DDIM   = 512;
constexpr int K2     = 1024;   // 2 * D (real|imag concatenated)
constexpr int NKEYS  = 8192;
constexpr int TOPK   = 64;
constexpr float EPSC   = 1e-8f;
constexpr float LN_EPSC = 1e-5f;

constexpr int BM = 64;   // block tile M
constexpr int BN = 128;  // block tile N
constexpr int BK = 32;   // k-chunk (bf16 wmma K)

// ---------------------------------------------------------------- helpers
__device__ inline v8f v8f_zero() {
  v8f z;
#pragma unroll
  for (int i = 0; i < 8; ++i) z[i] = 0.0f;
  return z;
}

__device__ inline v8f wmma_bf16(v16bf a, v16bf b, v8f c) {
  // D = A(16x32 bf16) * B(32x16 bf16) + C(f32)
  return __builtin_amdgcn_wmma_f32_16x16x32_bf16(
      /*neg_a=*/false, a, /*neg_b=*/false, b,
      /*c_mod=*/(short)0, c, /*reuse_a=*/false, /*reuse_b=*/false);
}

// Generic pointer -> 32-bit LDS byte address (aperture low bits).
__device__ inline unsigned lds_addr32(const void* p) {
  return (unsigned)(uintptr_t)p;
}

// CDNA5 async global->LDS copy, 16B per lane, tracked with ASYNCcnt.
__device__ inline void async_copy_b128(unsigned lds_off, const void* gptr) {
  unsigned long long ga = (unsigned long long)(uintptr_t)gptr;
  asm volatile("global_load_async_to_lds_b128 %0, %1, off"
               :: "v"(lds_off), "v"(ga) : "memory");
}

__device__ inline void wait_async0() {
  asm volatile("s_wait_asynccnt 0x0" ::: "memory");
}

// A fragment: 16x32 bf16 tile, LDS row-major [m][k], ldk = BK.
// lane L: m = L&15, h = L>>4;  element e -> k = (e&7) + 16*(e>>3) + 8*h
__device__ inline v16bf load_frag_A(const bf16_t* tile, int m0, int ldk) {
  const int lane = threadIdx.x & 31;
  const int m = lane & 15, h = lane >> 4;
  const bf16_t* row = tile + (m0 + m) * ldk;
  v8bf lo = *(const v8bf*)(row + 8 * h);        // k = 8h .. 8h+7
  v8bf hi = *(const v8bf*)(row + 16 + 8 * h);   // k = 16+8h .. 16+8h+7
  v16bf f;
#pragma unroll
  for (int e = 0; e < 8; ++e) { f[e] = lo[e]; f[8 + e] = hi[e]; }
  return f;
}

// B fragment: 32x16 bf16 tile, stored n-major in LDS: [n][k], ldk = BK.
// lane L: n = L&15, h = L>>4;  element e -> k = (e&7) + 16*(e>>3) + 8*h
__device__ inline v16bf load_frag_B(const bf16_t* tile, int n0, int ldk) {
  const int lane = threadIdx.x & 31;
  const int n = lane & 15, h = lane >> 4;
  const bf16_t* p = tile + (n0 + n) * ldk + 8 * h;
  v8bf lo = *(const v8bf*)p;
  v8bf hi = *(const v8bf*)(p + 16);
  v16bf f;
#pragma unroll
  for (int e = 0; e < 8; ++e) { f[e] = lo[e]; f[8 + e] = hi[e]; }
  return f;
}

__device__ inline float block_reduce_sum(float v, float* sbuf) {
  const int tid = threadIdx.x;
  sbuf[tid] = v;
  __syncthreads();
#pragma unroll
  for (int s = 128; s > 0; s >>= 1) {
    if (tid < s) sbuf[tid] += sbuf[tid + s];
    __syncthreads();
  }
  float r = sbuf[0];
  __syncthreads();
  return r;
}

__device__ inline float block_reduce_max(float v, float* sbuf) {
  const int tid = threadIdx.x;
  sbuf[tid] = v;
  __syncthreads();
#pragma unroll
  for (int s = 128; s > 0; s >>= 1) {
    if (tid < s) sbuf[tid] = fmaxf(sbuf[tid], sbuf[tid + s]);
    __syncthreads();
  }
  float r = sbuf[0];
  __syncthreads();
  return r;
}

// ---------------------------------------------------------------- conversions
// src row: [1024] fp32 interleaved (d,p) -> dst row: [1024] bf16 cat (r|i)
__global__ void convert_cat_kernel(const float* __restrict__ src,
                                   bf16_t* __restrict__ dst) {
  const int row = blockIdx.x, tid = threadIdx.x;
  const float4* p4 = (const float4*)(src + (size_t)row * K2);
  float4 v = p4[tid];   // (d=2t,p0) (d=2t,p1) (d=2t+1,p0) (d=2t+1,p1)
  bf16_t* d = dst + (size_t)row * K2;
  v2bf pr = {(bf16_t)v.x, (bf16_t)v.z};
  v2bf pi = {(bf16_t)v.y, (bf16_t)v.w};
  *(v2bf*)&d[2 * tid] = pr;
  *(v2bf*)&d[DDIM + 2 * tid] = pi;
}

// keys: [N][D][2] fp32 -> keysBT [n][1024] bf16 cat, plus k_mag[n]
__global__ void prep_keys_kernel(const float* __restrict__ keys,
                                 bf16_t* __restrict__ keysBT,
                                 float* __restrict__ kmag) {
  const int n = blockIdx.x, tid = threadIdx.x;
  const float4* p4 = (const float4*)(keys + (size_t)n * K2);
  float4 v = p4[tid];
  bf16_t* d = keysBT + (size_t)n * K2;
  v2bf pr = {(bf16_t)v.x, (bf16_t)v.z};
  v2bf pi = {(bf16_t)v.y, (bf16_t)v.w};
  *(v2bf*)&d[2 * tid] = pr;
  *(v2bf*)&d[DDIM + 2 * tid] = pi;
  __shared__ float red[256];
  float s = v.x * v.x + v.y * v.y + v.z * v.z + v.w * v.w;
  float tot = block_reduce_sum(s, red);
  if (tid == 0) kmag[n] = sqrtf(tot + EPSC);
}

// 512x512 fp32 -> transposed bf16 [j][k]
__global__ void conv_wT_kernel(const float* __restrict__ in,
                               bf16_t* __restrict__ outT) {
  __shared__ float tile[32][33];
  const int k0 = blockIdx.x * 32, j0 = blockIdx.y * 32;
  for (int i = threadIdx.y; i < 32; i += 8)
    tile[i][threadIdx.x] = in[(size_t)(k0 + i) * DDIM + j0 + threadIdx.x];
  __syncthreads();
  for (int i = threadIdx.y; i < 32; i += 8)
    outT[(size_t)(j0 + i) * DDIM + k0 + threadIdx.x] =
        (bf16_t)tile[threadIdx.x][i];
}

// q_mag from bf16 cat rows
__global__ void qmag_kernel(const bf16_t* __restrict__ qcat,
                            float* __restrict__ qmag) {
  const int row = blockIdx.x, tid = threadIdx.x;
  const bf16_t* rp = qcat + (size_t)row * K2;
  float s = 0.f;
  for (int i = tid; i < K2; i += 256) {
    float f = (float)rp[i];
    s += f * f;
  }
  __shared__ float red[256];
  float tot = block_reduce_sum(s, red);
  if (tid == 0) qmag[row] = sqrtf(tot + EPSC);
}

// ---------------------------------------------------------------- complex GEMM
// OR = Xr@Wr - Xi@Wi + br ; OI = Xr@Wi + Xi@Wr + bi
// A: [M][1024] bf16 cat.  WrT/WiT: [512][512] bf16 (row j, col k).
// WMODE 0: write bf16 cat [M][1024].  WMODE 1: write fp32 interleaved.
template <int WMODE>
__global__ __launch_bounds__(256) void phase_gemm_kernel(
    const bf16_t* __restrict__ A, const bf16_t* __restrict__ WrT,
    const bf16_t* __restrict__ WiT, const float* __restrict__ bias,
    bf16_t* __restrict__ outCat, float* __restrict__ outF32) {
  __shared__ __align__(16) bf16_t Asr[2][BM * BK];
  __shared__ __align__(16) bf16_t Asi[2][BM * BK];
  __shared__ __align__(16) bf16_t Bsr[2][BN * BK];
  __shared__ __align__(16) bf16_t Bsi[2][BN * BK];
  const int tid = threadIdx.x;
  const int row0 = blockIdx.y * BM;
  const int col0 = blockIdx.x * BN;
  const int wave = tid >> 5;
  const int wm = wave & 1;   // 0..1  (M groups of 32)
  const int wn = wave >> 1;  // 0..3  (N groups of 32)

  v8f arr[2][2], aii[2][2], aoi[2][2];
#pragma unroll
  for (int mi = 0; mi < 2; ++mi)
#pragma unroll
    for (int ni = 0; ni < 2; ++ni) {
      arr[mi][ni] = v8f_zero();
      aii[mi][ni] = v8f_zero();
      aoi[mi][ni] = v8f_zero();
    }

  // async stage of one k-chunk into buffer `bf`
  auto stage = [&](int bf, int k0) {
    {  // A: 256 16B units per plane -> 1 per thread per plane
      const int r = tid >> 2, c = (tid & 3) * 8;
      async_copy_b128(lds_addr32(&Asr[bf][r * BK + c]),
                      &A[(size_t)(row0 + r) * K2 + k0 + c]);
      async_copy_b128(lds_addr32(&Asi[bf][r * BK + c]),
                      &A[(size_t)(row0 + r) * K2 + DDIM + k0 + c]);
    }
#pragma unroll
    for (int t = 0; t < 2; ++t) {  // B: 512 units per plane -> 2 per thread
      const int u = tid + t * 256;
      const int r = u >> 2, c = (u & 3) * 8;
      async_copy_b128(lds_addr32(&Bsr[bf][r * BK + c]),
                      &WrT[(size_t)(col0 + r) * DDIM + k0 + c]);
      async_copy_b128(lds_addr32(&Bsi[bf][r * BK + c]),
                      &WiT[(size_t)(col0 + r) * DDIM + k0 + c]);
    }
  };

  stage(0, 0);
  int buf = 0;
  for (int k0 = 0; k0 < DDIM; k0 += BK) {
    // wait for the chunk we are about to consume (issued last iteration),
    // publish it to all waves
    wait_async0();
    __syncthreads();
    // issue the next chunk's loads NOW so the WMMAs below hide their latency
    if (k0 + BK < DDIM) stage(buf ^ 1, k0 + BK);

    v16bf ar[2], ai[2], br2[2], bi2[2];
#pragma unroll
    for (int mi = 0; mi < 2; ++mi) {
      ar[mi] = load_frag_A(Asr[buf], wm * 32 + mi * 16, BK);
      ai[mi] = load_frag_A(Asi[buf], wm * 32 + mi * 16, BK);
    }
#pragma unroll
    for (int ni = 0; ni < 2; ++ni) {
      br2[ni] = load_frag_B(Bsr[buf], wn * 32 + ni * 16, BK);
      bi2[ni] = load_frag_B(Bsi[buf], wn * 32 + ni * 16, BK);
    }
#pragma unroll
    for (int mi = 0; mi < 2; ++mi)
#pragma unroll
      for (int ni = 0; ni < 2; ++ni) {
        arr[mi][ni] = wmma_bf16(ar[mi], br2[ni], arr[mi][ni]);
        aii[mi][ni] = wmma_bf16(ai[mi], bi2[ni], aii[mi][ni]);
        aoi[mi][ni] = wmma_bf16(ar[mi], bi2[ni], aoi[mi][ni]);
        aoi[mi][ni] = wmma_bf16(ai[mi], br2[ni], aoi[mi][ni]);
      }
    buf ^= 1;
  }

  const int lane = tid & 31;
  const int cn = lane & 15, ch = lane >> 4;
#pragma unroll
  for (int mi = 0; mi < 2; ++mi)
#pragma unroll
    for (int ni = 0; ni < 2; ++ni) {
      const int jj = col0 + wn * 32 + ni * 16 + cn;
      const float br_ = bias[jj * 2 + 0], bi_ = bias[jj * 2 + 1];
#pragma unroll
      for (int r = 0; r < 8; ++r) {
        const int m = row0 + wm * 32 + mi * 16 + (r + 8 * ch);
        const float orv = arr[mi][ni][r] - aii[mi][ni][r] + br_;
        const float oiv = aoi[mi][ni][r] + bi_;
        if (WMODE == 0) {
          outCat[(size_t)m * K2 + jj] = (bf16_t)orv;
          outCat[(size_t)m * K2 + DDIM + jj] = (bf16_t)oiv;
        } else {
          outF32[(size_t)m * K2 + jj * 2 + 0] = orv;
          outF32[(size_t)m * K2 + jj * 2 + 1] = oiv;
        }
      }
    }
}

// ---------------------------------------------------------------- coherence GEMM
// coh[m][n] = (sum_k A[m][k]*Bt[n][k]) / (qmag[m]*kmag[n] + eps)
__global__ __launch_bounds__(256) void coh_gemm_kernel(
    const bf16_t* __restrict__ A, const bf16_t* __restrict__ Bt,
    const float* __restrict__ qmag, const float* __restrict__ kmag,
    float* __restrict__ coh) {
  __shared__ __align__(16) bf16_t As[2][BM * BK];
  __shared__ __align__(16) bf16_t Bs[2][BN * BK];
  const int tid = threadIdx.x;
  const int row0 = blockIdx.y * BM;
  const int col0 = blockIdx.x * BN;
  const int wave = tid >> 5;
  const int wm = wave & 1;
  const int wn = wave >> 1;

  v8f acc[2][2];
#pragma unroll
  for (int mi = 0; mi < 2; ++mi)
#pragma unroll
    for (int ni = 0; ni < 2; ++ni) acc[mi][ni] = v8f_zero();

  auto stage = [&](int bf, int k0) {
    {
      const int r = tid >> 2, c = (tid & 3) * 8;
      async_copy_b128(lds_addr32(&As[bf][r * BK + c]),
                      &A[(size_t)(row0 + r) * K2 + k0 + c]);
    }
#pragma unroll
    for (int t = 0; t < 2; ++t) {
      const int u = tid + t * 256;
      const int r = u >> 2, c = (u & 3) * 8;
      async_copy_b128(lds_addr32(&Bs[bf][r * BK + c]),
                      &Bt[(size_t)(col0 + r) * K2 + k0 + c]);
    }
  };

  stage(0, 0);
  int buf = 0;
  for (int k0 = 0; k0 < K2; k0 += BK) {
    wait_async0();
    __syncthreads();
    if (k0 + BK < K2) {
      stage(buf ^ 1, k0 + BK);
      // hint the chunk after next toward L2
      if (k0 + 2 * BK < K2)
        __builtin_prefetch(&Bt[(size_t)(col0 + (tid >> 1)) * K2 + k0 + 2 * BK],
                           0, 1);
    }

    v16bf af[2], bfv[2];
#pragma unroll
    for (int mi = 0; mi < 2; ++mi)
      af[mi] = load_frag_A(As[buf], wm * 32 + mi * 16, BK);
#pragma unroll
    for (int ni = 0; ni < 2; ++ni)
      bfv[ni] = load_frag_B(Bs[buf], wn * 32 + ni * 16, BK);
#pragma unroll
    for (int mi = 0; mi < 2; ++mi)
#pragma unroll
      for (int ni = 0; ni < 2; ++ni)
        acc[mi][ni] = wmma_bf16(af[mi], bfv[ni], acc[mi][ni]);

    buf ^= 1;
  }

  const int lane = tid & 31;
  const int cn = lane & 15, ch = lane >> 4;
#pragma unroll
  for (int mi = 0; mi < 2; ++mi)
#pragma unroll
    for (int ni = 0; ni < 2; ++ni) {
      const int jj = col0 + wn * 32 + ni * 16 + cn;
      const float km = kmag[jj];
#pragma unroll
      for (int r = 0; r < 8; ++r) {
        const int m = row0 + wm * 32 + mi * 16 + (r + 8 * ch);
        coh[(size_t)m * NKEYS + jj] = acc[mi][ni][r] / (qmag[m] * km + EPSC);
      }
    }
}

// ---------------------------------------------------------------- top-k + softmax
__global__ __launch_bounds__(256) void topk_softmax_kernel(
    float* __restrict__ attn, int* __restrict__ tidx, float* __restrict__ tw) {
  const int row = blockIdx.x, tid = threadIdx.x;
  float* rp = attn + (size_t)row * NKEYS;
  __shared__ unsigned su[NKEYS];  // 32 KB row cache (sortable uints)
  __shared__ unsigned hist[256];
  __shared__ float red[256];
  __shared__ unsigned sh_prefix;
  __shared__ int sh_k;
  __shared__ int cnt;

  for (int i = tid; i < NKEYS; i += 256) {
    unsigned u = __float_as_uint(rp[i]);
    su[i] = (u & 0x80000000u) ? ~u : (u | 0x80000000u);
  }
  if (tid == 0) { sh_prefix = 0u; sh_k = TOPK; cnt = 0; }
  __syncthreads();

  unsigned pmask = 0u;
#pragma unroll
  for (int pass = 0; pass < 4; ++pass) {
    const int shift = 24 - 8 * pass;
    hist[tid] = 0u;
    __syncthreads();
    const unsigned prefix = sh_prefix;
    for (int i = tid; i < NKEYS; i += 256) {
      const unsigned u = su[i];
      if ((u & pmask) == prefix) atomicAdd(&hist[(u >> shift) & 255u], 1u);
    }
    __syncthreads();
    if (tid == 0) {
      int need = sh_k;
      unsigned run = 0;
      for (int b = 255; b >= 0; --b) {
        const unsigned h = hist[b];
        if (run + h >= (unsigned)need) {
          sh_prefix = prefix | ((unsigned)b << shift);
          sh_k = need - (int)run;
          break;
        }
        run += h;
      }
    }
    __syncthreads();
    pmask |= (0xFFu << shift);
  }
  const unsigned thr = sh_prefix;  // exact bit pattern of 64th-largest

  float lmax = -3.4e38f;
  for (int i = tid; i < NKEYS; i += 256)
    if (su[i] >= thr) lmax = fmaxf(lmax, rp[i]);
  const float m = block_reduce_max(lmax, red);

  float lsum = 0.f;
  for (int i = tid; i < NKEYS; i += 256)
    if (su[i] >= thr) lsum += __expf(rp[i] - m);
  const float tot = block_reduce_sum(lsum, red);
  const float inv = 1.0f / tot;

  for (int i = tid; i < NKEYS; i += 256) {
    float a = 0.f;
    if (su[i] >= thr) {
      a = __expf(rp[i] - m) * inv;
      const int p = atomicAdd(&cnt, 1);
      if (p < TOPK) {
        tidx[(size_t)row * TOPK + p] = i;
        tw[(size_t)row * TOPK + p] = a;
      }
    }
    rp[i] = a;
  }
}

// ---------------------------------------------------------------- sparse retrieve + LN
__global__ __launch_bounds__(256) void retrieve_ln_kernel(
    const float* __restrict__ values, const int* __restrict__ tidx,
    const float* __restrict__ tw, const float* __restrict__ gamma,
    const float* __restrict__ beta, bf16_t* __restrict__ xcat) {
  const int row = blockIdx.x, tid = threadIdx.x;
  __shared__ float sw[TOPK];
  __shared__ int sidx[TOPK];
  __shared__ float x[K2];
  __shared__ float mag[DDIM];
  __shared__ float red[256];
  if (tid < TOPK) {
    sw[tid] = tw[(size_t)row * TOPK + tid];
    sidx[tid] = tidx[(size_t)row * TOPK + tid];
  }
  __syncthreads();

  float4 acc = {0.f, 0.f, 0.f, 0.f};
#pragma unroll 4
  for (int j = 0; j < TOPK; ++j) {
    const float w = sw[j];
    const float4 v = ((const float4*)(values + (size_t)sidx[j] * K2))[tid];
    acc.x += w * v.x; acc.y += w * v.y; acc.z += w * v.z; acc.w += w * v.w;
  }
  ((float4*)x)[tid] = acc;
  __syncthreads();

  float lsum = 0.f, lsq = 0.f;
  for (int d = tid; d < DDIM; d += 256) {
    const float xr = x[2 * d], xi = x[2 * d + 1];
    const float mg = sqrtf(xr * xr + xi * xi + EPSC);
    mag[d] = mg;
    lsum += mg;
    lsq += mg * mg;
  }
  const float tsum = block_reduce_sum(lsum, red);
  const float tsq = block_reduce_sum(lsq, red);
  const float mu = tsum / (float)DDIM;
  const float var = tsq / (float)DDIM - mu * mu;
  const float inv = rsqrtf(var + LN_EPSC);
  for (int d = tid; d < DDIM; d += 256) {
    const float mg = mag[d];
    const float nm = (mg - mu) * inv * gamma[d] + beta[d];
    const float sc = nm / mg;
    xcat[(size_t)row * K2 + d] = (bf16_t)(x[2 * d] * sc);
    xcat[(size_t)row * K2 + DDIM + d] = (bf16_t)(x[2 * d + 1] * sc);
  }
}

// ---------------------------------------------------------------- launch
extern "C" void kernel_launch(void* const* d_in, const int* in_sizes, int n_in,
                              void* d_out, int out_size, void* d_ws,
                              size_t ws_size, hipStream_t stream) {
  (void)in_sizes; (void)n_in; (void)out_size; (void)ws_size;
  const float* query = (const float*)d_in[0];
  const float* keys  = (const float*)d_in[1];
  const float* values = (const float*)d_in[2];
  const float* qp_wr = (const float*)d_in[3];
  const float* qp_wi = (const float*)d_in[4];
  const float* qp_b  = (const float*)d_in[5];
  const float* op_wr = (const float*)d_in[6];
  const float* op_wi = (const float*)d_in[7];
  const float* op_b  = (const float*)d_in[8];
  const float* ln_g  = (const float*)d_in[9];
  const float* ln_b  = (const float*)d_in[10];

  char* ws = (char*)d_ws;
  size_t off = 0;
  auto alloc = [&](size_t bytes) -> void* {
    off = (off + 255) & ~(size_t)255;
    void* p = ws + off;
    off += bytes;
    return p;
  };
  bf16_t* qcat_in  = (bf16_t*)alloc((size_t)BSROWS * K2 * 2);
  bf16_t* qcat_mid = (bf16_t*)alloc((size_t)BSROWS * K2 * 2);
  bf16_t* xcat_ln  = (bf16_t*)alloc((size_t)BSROWS * K2 * 2);
  bf16_t* keysBT   = (bf16_t*)alloc((size_t)NKEYS * K2 * 2);
  bf16_t* wrT_qp   = (bf16_t*)alloc((size_t)DDIM * DDIM * 2);
  bf16_t* wiT_qp   = (bf16_t*)alloc((size_t)DDIM * DDIM * 2);
  bf16_t* wrT_op   = (bf16_t*)alloc((size_t)DDIM * DDIM * 2);
  bf16_t* wiT_op   = (bf16_t*)alloc((size_t)DDIM * DDIM * 2);
  float*  qmag     = (float*)alloc((size_t)BSROWS * 4);
  float*  kmag     = (float*)alloc((size_t)NKEYS * 4);
  int*    tidx     = (int*)alloc((size_t)BSROWS * TOPK * 4);
  float*  tw       = (float*)alloc((size_t)BSROWS * TOPK * 4);

  float* outF = (float*)d_out;                 // [4096][512][2]
  float* attn = outF + (size_t)BSROWS * K2;    // [4096][8192]

  // 1) conversions
  convert_cat_kernel<<<BSROWS, 256, 0, stream>>>(query, qcat_in);
  prep_keys_kernel<<<NKEYS, 256, 0, stream>>>(keys, keysBT, kmag);
  dim3 tg(16, 16), tb(32, 8);
  conv_wT_kernel<<<tg, tb, 0, stream>>>(qp_wr, wrT_qp);
  conv_wT_kernel<<<tg, tb, 0, stream>>>(qp_wi, wiT_qp);
  conv_wT_kernel<<<tg, tb, 0, stream>>>(op_wr, wrT_op);
  conv_wT_kernel<<<tg, tb, 0, stream>>>(op_wi, wiT_op);

  // 2) query phase projection (WMMA bf16, async double-buffered LDS)
  phase_gemm_kernel<0><<<dim3(DDIM / BN, BSROWS / BM), 256, 0, stream>>>(
      qcat_in, wrT_qp, wiT_qp, qp_b, qcat_mid, nullptr);

  // 3) q magnitudes
  qmag_kernel<<<BSROWS, 256, 0, stream>>>(qcat_mid, qmag);

  // 4) coherence GEMM (WMMA bf16) -> attn region of d_out
  coh_gemm_kernel<<<dim3(NKEYS / BN, BSROWS / BM), 256, 0, stream>>>(
      qcat_mid, keysBT, qmag, kmag, attn);

  // 5) exact top-64 radix select + masked softmax (in place) + compaction
  topk_softmax_kernel<<<BSROWS, 256, 0, stream>>>(attn, tidx, tw);

  // 6) sparse retrieve (L2-resident gather) fused with magnitude LayerNorm
  retrieve_ln_kernel<<<BSROWS, 256, 0, stream>>>(values, tidx, tw, ln_g, ln_b,
                                                 xcat_ln);

  // 7) output phase projection (WMMA bf16) -> fp32 interleaved d_out
  phase_gemm_kernel<1><<<dim3(DDIM / BN, BSROWS / BM), 256, 0, stream>>>(
      xcat_ln, wrT_op, wiT_op, op_b, nullptr, outF);
}
